// RelativeMultiHeadAttention_38208029066046
// MI455X (gfx1250) — matrix operand order repaired and checked by reading, not compiled, not run on hardware
//
#include <hip/hip_runtime.h>
#include <cstdint>

// ---------------------------------------------------------------------------
// Transformer-XL Relative Multi-Head Attention for gfx1250 (MI455X)
// B=2, S=2048, D_MODEL=512, H=8, Dh=64.
// GEMMs: bf16 WMMA + TDM (tensor_load_to_lds) double-buffered LDS staging.
// Attention: fused 16-row strips in LDS, WMMA for QK^T, QP^T, attn@V.
// ---------------------------------------------------------------------------

#define SLEN   2048
#define DMODEL 512
#define NHEAD  8
#define DHEAD  64
#define NROWS  4096            // B*S
#define NELEM  (NROWS * DMODEL)

typedef __attribute__((ext_vector_type(16))) __bf16        v16bf;
typedef __attribute__((ext_vector_type(8)))  float         v8f;
typedef __attribute__((ext_vector_type(4)))  unsigned int  v4u;
typedef __attribute__((ext_vector_type(8)))  int           v8i;
typedef __attribute__((ext_vector_type(4)))  int           v4i;

__device__ __forceinline__ float toF(float x) { return x; }
__device__ __forceinline__ float toF(__bf16 x) { return (float)x; }

// ---------------------------------------------------------------------------
// Tensor Data Mover descriptor (D#) for a 2D tile load, per CDNA5 ISA 8.3/8.4.
// All dims/strides in data_size units; addresses in bytes.
// ---------------------------------------------------------------------------
struct TdmDesc { v4u g0; v8i g1; };

__device__ __forceinline__ TdmDesc make_tdm_2d(
    unsigned ldsOff, const void* gptr,
    unsigned tileX, unsigned tileY, unsigned strideElems,
    unsigned dsizeCode, unsigned padInterval, unsigned padAmount, bool padEn)
{
    const unsigned long long ga = (unsigned long long)(uintptr_t)gptr;
    TdmDesc d;
    d.g0[0] = 1u;                                   // count=1 valid descriptor
    d.g0[1] = ldsOff;                               // lds_addr (bytes)
    d.g0[2] = (unsigned)ga;                         // global_addr[31:0]
    d.g0[3] = (unsigned)((ga >> 32) & 0x01ffffffu)  // global_addr[56:32]
            | (2u << 30);                           // type = 2 ("image")
    unsigned f = (dsizeCode & 3u) << 16;            // data_size; workgroup_mask=0
    if (padEn)
        f |= (1u << 20)                             // pad_enable
           | ((padInterval & 7u) << 22)             // pad_interval (log2 dwords - 1)
           | ((padAmount & 127u) << 25);            // pad_amount (dwords - 1)
    d.g1[0] = (int)f;
    d.g1[1] = (int)((tileX & 0xffffu) << 16);                               // tensor_dim0 lo
    d.g1[2] = (int)(((tileX >> 16) & 0xffffu) | ((tileY & 0xffffu) << 16)); // dim0 hi | dim1 lo
    d.g1[3] = (int)(((tileY >> 16) & 0xffffu) | ((tileX & 0xffffu) << 16)); // dim1 hi | tile_dim0
    d.g1[4] = (int)(tileY & 0xffffu);                                       // tile_dim1 | tile_dim2=0
    d.g1[5] = (int)strideElems;                                             // dim0_stride[31:0]
    d.g1[6] = 0;                                                            // stride hi | dim1_stride lo
    d.g1[7] = 0;
    return d;
}

__device__ __forceinline__ void tdm_issue(const TdmDesc& d)
{
    const v4i z4 = {0, 0, 0, 0};
#if __clang_major__ >= 23
    const v8i z8 = {0, 0, 0, 0, 0, 0, 0, 0};
    __builtin_amdgcn_tensor_load_to_lds(d.g0, d.g1, z4, z4, z8, 0);
#else
    __builtin_amdgcn_tensor_load_to_lds(d.g0, d.g1, z4, z4, 0);
#endif
}

// ===========================================================================
// out = A[4096,512] @ W[512,512] (+bias) (+col-add)  via bf16 WMMA.
// LDS tiles staged by the Tensor Data Mover, double-buffered:
//   A tile 64x32 fp32 (pitch 34 via TDM pad), W tile 32x128 fp32 (pitch 130).
// For AT=__bf16 (context) the A tile is staged manually (TDM can't convert).
// Workgroup 256 thr (8 waves) -> 64 rows x 128 cols; wave -> 16x64 (4 WMMA acc).
// ===========================================================================
#define APITCH 34
#define WPITCH 130
#define A_FLOATS (64 * APITCH)          // 2176
#define W_FLOATS (32 * WPITCH)          // 4160
#define OFF_A0 0u
#define OFF_W0 ((unsigned)(2 * A_FLOATS))

template <typename AT, bool OUTF>
__global__ __launch_bounds__(256) void gemm_xw(
    const AT* __restrict__ A, const float* __restrict__ W,
    const float* __restrict__ bias,
    const float* __restrict__ cadd0, const float* __restrict__ cadd1,
    __bf16* __restrict__ out0, __bf16* __restrict__ out1,
    float* __restrict__ outf)
{
    constexpr bool ATDM = (sizeof(AT) == 4);        // float A -> TDM; bf16 -> manual

    __shared__ float smem[2 * A_FLOATS + 2 * W_FLOATS];   // ~50.7 KB, offset 0 in LDS

    const int t    = threadIdx.x;
    const int lane = t & 31;
    const int wv   = t >> 5;
    const int mSub = wv & 3;
    const int nSub = wv >> 2;
    const int lm   = lane & 15;
    const int hi   = lane >> 4;
    const int rowBase = blockIdx.x * 64;
    const int colBase = blockIdx.y * 128;

    auto issueTile = [&](int kt2) {
        const unsigned bp = (unsigned)(kt2 & 1);
        if (ATDM) {
            TdmDesc dA = make_tdm_2d(
                (OFF_A0 + bp * (unsigned)A_FLOATS) * 4u,
                (const char*)A + ((size_t)rowBase * DMODEL + (size_t)kt2 * 32) * sizeof(AT),
                /*tileX*/32u, /*tileY*/64u, /*stride*/DMODEL,
                /*dsize 4B*/2u, /*pad every 32 dw*/4u, /*pad 2 dw*/1u, true);
            tdm_issue(dA);
        }
        TdmDesc dW = make_tdm_2d(
            (OFF_W0 + bp * (unsigned)W_FLOATS) * 4u,
            (const char*)W + ((size_t)(kt2 * 32) * DMODEL + colBase) * sizeof(float),
            /*tileX*/128u, /*tileY*/32u, /*stride*/DMODEL,
            2u, /*pad every 128 dw*/6u, /*pad 2 dw*/1u, true);
        tdm_issue(dW);
    };

    v8f acc[4];
    #pragma unroll
    for (int q = 0; q < 4; ++q)
        #pragma unroll
        for (int r = 0; r < 8; ++r) acc[q][r] = 0.0f;

    if (wv == 0) issueTile(0);

    for (int kt = 0; kt < DMODEL / 32; ++kt) {
        const int cur = kt & 1;
        if (wv == 0 && kt + 1 < DMODEL / 32) issueTile(kt + 1);
        if (!ATDM) {
            // manual bf16 -> fp32 staging of the A tile (single buffer)
            const int k0 = kt * 32;
            for (int x = t; x < 64 * 32; x += 256) {
                const int r = x >> 5, c = x & 31;
                smem[OFF_A0 + r * APITCH + c] = toF(A[(rowBase + r) * DMODEL + k0 + c]);
            }
        }
        if (wv == 0) {
            // wait until the CURRENT tile's TDM ops are done (in-order per wave);
            // builtin requires an integer constant argument -> literal branches.
            if (kt + 1 < DMODEL / 32) {
                if (ATDM) __builtin_amdgcn_s_wait_tensorcnt(2);
                else      __builtin_amdgcn_s_wait_tensorcnt(1);
            } else {
                __builtin_amdgcn_s_wait_tensorcnt(0);
            }
        }
        __syncthreads();                 // current tiles resident in LDS

        const float* sAp = smem + (ATDM ? (OFF_A0 + (unsigned)cur * A_FLOATS) : OFF_A0);
        const float* sWp = smem + OFF_W0 + (unsigned)cur * W_FLOATS;

        // A fragment: M=lm, K low-lanes {0..15}, high-lanes {16..31}
        v16bf af;
        {
            const int arow = mSub * 16 + lm;
            const int kb   = hi ? 16 : 0;
            #pragma unroll
            for (int i = 0; i < 8; ++i) {
                af[i]     = (__bf16)sAp[arow * APITCH + kb + i];
                af[i + 8] = (__bf16)sAp[arow * APITCH + kb + 8 + i];
            }
        }
        #pragma unroll
        for (int q = 0; q < 4; ++q) {
            v16bf bfg;
            const int col = nSub * 64 + q * 16 + lm;
            const int kb  = hi ? 16 : 0;
            #pragma unroll
            for (int i = 0; i < 16; ++i) bfg[i] = (__bf16)sWp[(kb + i) * WPITCH + col];
            acc[q] = __builtin_amdgcn_wmma_f32_16x16x32_bf16(
                false, af, false, bfg, (short)0, acc[q], false, false);
        }
        __syncthreads();                 // everyone done before buffers are reused
    }

    // epilogue
    #pragma unroll
    for (int q = 0; q < 4; ++q) {
        const int col  = colBase + nSub * 64 + q * 16 + lm;
        const float bb = bias ? bias[col] : 0.0f;
        const float c0 = cadd0 ? cadd0[col] : 0.0f;
        #pragma unroll
        for (int r = 0; r < 8; ++r) {
            const int row = rowBase + mSub * 16 + hi * 8 + r;
            const float v = acc[q][r] + bb;
            if (OUTF) {
                outf[row * DMODEL + col] = v + c0;
            } else {
                out0[row * DMODEL + col] = (__bf16)(v + c0);
                if (out1) out1[row * DMODEL + col] = (__bf16)(v + cadd1[col]);
            }
        }
    }
}

// ===========================================================================
// Fused relative attention.  One workgroup per (b, h, 16-row tile).
// ===========================================================================
__device__ __forceinline__ float shifted_pos(const float* sPd, const float* sPex,
                                             int r, int i, int j)
{
    if (j <= i)     return sPd[r * SLEN + (SLEN - 1 - i + j)];
    if (j == i + 1) return 0.0f;
    const int l = j - i - 2;
    return (r < 15) ? sPd[(r + 1) * SLEN + l] : sPex[l];
}

__global__ __launch_bounds__(256) void rel_attn(
    const __bf16* __restrict__ qu, const __bf16* __restrict__ qv,
    const __bf16* __restrict__ kk, const __bf16* __restrict__ vv,
    const __bf16* __restrict__ pp, const unsigned char* __restrict__ mask,
    __bf16* __restrict__ ctx)
{
    extern __shared__ float lds[];
    float* sC   = lds;                        // 16*2048 content -> exp(scores)
    float* sPd  = lds + 16 * SLEN;            // 16*2048 pos dots
    float* sPex = lds + 32 * SLEN;            // 2048 boundary row (reused as sCtx)
    float* sCtx = sPex;                       // 16*64
    float* sQx  = lds + 32 * SLEN + SLEN;     // 64 (q+v boundary row -> row sums)

    const int t    = threadIdx.x;
    const int lane = t & 31;
    const int wv   = t >> 5;
    const int lm   = lane & 15;
    const int hi   = lane >> 4;

    const int it = blockIdx.x & 127;
    const int h  = (blockIdx.x >> 7) & 7;
    const int b  = blockIdx.x >> 10;
    const int i0 = it * 16;
    const int rowOff = (b * SLEN) * DMODEL + h * DHEAD;
    const unsigned char* mb = mask + b * SLEN;
    const float SCALE = 0.044194173824159216f;            // 1/sqrt(512)

    // ---- phase 0: per-wave A fragments for (q+u), (q+v)
    v16bf aQu[2], aQv[2];
    {
        const int qrow = rowOff + (i0 + lm) * DMODEL;
        const int kb   = hi ? 16 : 0;
        #pragma unroll
        for (int ks = 0; ks < 2; ++ks) {
            const int d0 = ks * 32;
            #pragma unroll
            for (int i = 0; i < 8; ++i) {
                aQu[ks][i]     = qu[qrow + d0 + kb + i];
                aQu[ks][i + 8] = qu[qrow + d0 + kb + 8 + i];
                aQv[ks][i]     = qv[qrow + d0 + kb + i];
                aQv[ks][i + 8] = qv[qrow + d0 + kb + 8 + i];
            }
        }
    }
    if (t < DHEAD)
        sQx[t] = (it < 127) ? (float)qv[rowOff + (i0 + 16) * DMODEL + t] : 0.0f;
    __syncthreads();

    // ---- phase 1: content + pos strips via WMMA (wave owns 16 j-tiles)
    for (int ii = 0; ii < 16; ++ii) {
        const int j0 = (wv * 16 + ii) * 16;
        v8f cacc, pacc;
        #pragma unroll
        for (int r = 0; r < 8; ++r) { cacc[r] = 0.0f; pacc[r] = 0.0f; }
        #pragma unroll
        for (int ks = 0; ks < 2; ++ks) {
            const int kbB  = hi ? 16 : 0;
            const int krow = rowOff + (j0 + lm) * DMODEL + ks * 32 + kbB;
            v16bf bK, bP;
            #pragma unroll
            for (int i = 0; i < 16; ++i) { bK[i] = kk[krow + i]; bP[i] = pp[krow + i]; }
            cacc = __builtin_amdgcn_wmma_f32_16x16x32_bf16(
                false, aQu[ks], false, bK, (short)0, cacc, false, false);
            pacc = __builtin_amdgcn_wmma_f32_16x16x32_bf16(
                false, aQv[ks], false, bP, (short)0, pacc, false, false);
        }
        #pragma unroll
        for (int r = 0; r < 8; ++r) {
            const int m = r + hi * 8;
            sC [m * SLEN + j0 + lm] = cacc[r];
            sPd[m * SLEN + j0 + lm] = pacc[r];
        }
    }
    // boundary Pdot row i0+16
    for (int l = t; l < SLEN; l += 256) {
        const __bf16* prow = pp + rowOff + l * DMODEL;
        float acc = 0.0f;
        #pragma unroll
        for (int d = 0; d < DHEAD; ++d) acc += sQx[d] * (float)prow[d];
        sPex[l] = acc;
    }
    __syncthreads();

    // ---- phase 2: rel-shift assembly, mask, softmax (2 rows per wave)
    #pragma unroll
    for (int rr = 0; rr < 2; ++rr) {
        const int r = wv * 2 + rr;
        const int i = i0 + r;
        float mx = -3.0e38f;
        for (int j = lane; j < SLEN; j += 32) {
            float s = (sC[r * SLEN + j] + shifted_pos(sPd, sPex, r, i, j)) * SCALE;
            if (mb[j]) s = -1.0e9f;
            sC[r * SLEN + j] = s;
            mx = fmaxf(mx, s);
        }
        #pragma unroll
        for (int o = 16; o > 0; o >>= 1) mx = fmaxf(mx, __shfl_xor(mx, o));
        float sum = 0.0f;
        for (int j = lane; j < SLEN; j += 32) {
            const float e = exp2f((sC[r * SLEN + j] - mx) * 1.4426950408889634f);
            sC[r * SLEN + j] = e;
            sum += e;
        }
        #pragma unroll
        for (int o = 16; o > 0; o >>= 1) sum += __shfl_xor(sum, o);
        if (lane == 0) sQx[r] = sum;
    }
    __syncthreads();

    // ---- phase 3: context = attn @ V via WMMA + LDS reduction
    for (int x = t; x < 16 * DHEAD; x += 256) sCtx[x] = 0.0f;
    __syncthreads();
    {
        const int nq    = wv & 3;
        const int khalf = wv >> 2;
        v8f acc;
        #pragma unroll
        for (int r = 0; r < 8; ++r) acc[r] = 0.0f;
        for (int kt2 = 0; kt2 < 32; ++kt2) {
            const int k0 = (khalf * 32 + kt2) * 32;
            v16bf af;
            {
                const int kb = hi ? 16 : 0;
                #pragma unroll
                for (int i = 0; i < 8; ++i) {
                    af[i]     = (__bf16)sC[lm * SLEN + k0 + kb + i];
                    af[i + 8] = (__bf16)sC[lm * SLEN + k0 + kb + 8 + i];
                }
            }
            v16bf bfg;
            {
                const int kbB = hi ? 16 : 0;
                #pragma unroll
                for (int i = 0; i < 16; ++i)
                    bfg[i] = vv[rowOff + (k0 + kbB + i) * DMODEL + nq * 16 + lm];
            }
            acc = __builtin_amdgcn_wmma_f32_16x16x32_bf16(
                false, af, false, bfg, (short)0, acc, false, false);
        }
        #pragma unroll
        for (int r = 0; r < 8; ++r) {
            const int m = r + hi * 8;
            atomicAdd(&sCtx[m * DHEAD + nq * 16 + lm], acc[r]);
        }
    }
    __syncthreads();

    // ---- phase 4: normalize and write context (bf16)
    for (int x = t; x < 16 * DHEAD; x += 256) {
        const int m = x >> 6, c = x & 63;
        ctx[rowOff + (i0 + m) * DMODEL + c] = (__bf16)(sCtx[m * DHEAD + c] / sQx[m]);
    }
}

// ===========================================================================
extern "C" void kernel_launch(void* const* d_in, const int* in_sizes, int n_in,
                              void* d_out, int out_size, void* d_ws, size_t ws_size,
                              hipStream_t stream)
{
    (void)in_sizes; (void)n_in; (void)out_size; (void)ws_size;
    const float* query  = (const float*)d_in[0];
    const float* key    = (const float*)d_in[1];
    const float* value  = (const float*)d_in[2];
    const float* posemb = (const float*)d_in[3];
    const unsigned char* mask = (const unsigned char*)d_in[4];
    const float* Wq = (const float*)d_in[5];
    const float* bq = (const float*)d_in[6];
    const float* Wk = (const float*)d_in[7];
    const float* bk = (const float*)d_in[8];
    const float* Wv = (const float*)d_in[9];
    const float* bv = (const float*)d_in[10];
    const float* Wp = (const float*)d_in[11];
    const float* Wo = (const float*)d_in[12];
    const float* bo = (const float*)d_in[13];
    const float* u_bias = (const float*)d_in[14];   // [8,64] -> flat [512]
    const float* v_bias = (const float*)d_in[15];

    __bf16* base = (__bf16*)d_ws;
    __bf16* quW  = base;
    __bf16* qvW  = base + 1 * (size_t)NELEM;
    __bf16* kW   = base + 2 * (size_t)NELEM;
    __bf16* vW   = base + 3 * (size_t)NELEM;
    __bf16* pW   = base + 4 * (size_t)NELEM;
    __bf16* ctxW = base + 5 * (size_t)NELEM;

    const dim3 gG(NROWS / 64, DMODEL / 128), blk(256);

    gemm_xw<float, false><<<gG, blk, 0, stream>>>(query,  Wq, bq, u_bias, v_bias,
                                                  quW, qvW, nullptr);
    gemm_xw<float, false><<<gG, blk, 0, stream>>>(key,    Wk, bk, nullptr, nullptr,
                                                  kW, nullptr, nullptr);
    gemm_xw<float, false><<<gG, blk, 0, stream>>>(value,  Wv, bv, nullptr, nullptr,
                                                  vW, nullptr, nullptr);
    gemm_xw<float, false><<<gG, blk, 0, stream>>>(posemb, Wp, nullptr, nullptr, nullptr,
                                                  pW, nullptr, nullptr);

    const size_t attnLds = (size_t)(32 * SLEN + SLEN + 64) * sizeof(float); // ~264 KB
    rel_attn<<<dim3(2 * NHEAD * (SLEN / 16)), blk, attnLds, stream>>>(
        quW, qvW, kW, vW, pW, mask, ctxW);

    gemm_xw<__bf16, true><<<gG, blk, 0, stream>>>(ctxW, Wo, bo, nullptr, nullptr,
                                                  nullptr, nullptr, (float*)d_out);
}
